// MLGrouping_24790551232813
// MI455X (gfx1250) — matrix operand order repaired and checked by reading, not compiled, not run on hardware
//
#include <hip/hip_runtime.h>
#include <hip/hip_bf16.h>

typedef __attribute__((ext_vector_type(16))) _Float16 v16h;
typedef __attribute__((ext_vector_type(8)))  _Float16 v8h;
typedef __attribute__((ext_vector_type(8)))  float    v8f;
typedef __attribute__((ext_vector_type(4)))  float    v4f;

#define NSEG   4096
#define IN_DIM 256
#define NCOL   512      // K*D = 8*64
#define MTILE  32

// ---------------- workspace zeroing ----------------
__global__ void zero_ws(float* __restrict__ p, int n) {
  int i = blockIdx.x * 256 + threadIdx.x;
  if (i < n) p[i] = 0.f;
}

// ---------------- pack W[k][i][d] (f32) into f16 WMMA-B fragments ----------
// Fragment layout (wave32 B, 32x16 f16 tile): lane L holds column (L&15),
// 16 contiguous halves h -> inner dim i = 32*t + (L>>4)*16 + h.
// Storage: frag[((c*8 + t)*32 + lane)*16 + h], c = column-tile (0..31).
__global__ void pack_w(const float* __restrict__ W, _Float16* __restrict__ frag) {
  int tid = blockIdx.x * 256 + threadIdx.x;          // (c*8 + t)*32 + lane
  if (tid >= 32 * 8 * 32) return;
  int lane = tid & 31;
  int t    = (tid >> 5) & 7;
  int c    = tid >> 8;
  int col  = (c << 4) + (lane & 15);
  int k = col >> 6, d = col & 63;
  int ibase = (t << 5) + ((lane >> 4) << 4);
  _Float16* o = frag + (size_t)tid * 16;
  #pragma unroll
  for (int h = 0; h < 16; ++h) {
    int i = ibase + h;
    o[h] = (_Float16)W[((size_t)k * IN_DIM + i) * 64 + d];
  }
}

// ---------------- per-segment counts via binary search (batch is sorted) ---
__global__ void seg_counts(const int* __restrict__ batch,
                           float* __restrict__ counts, int N) {
  int g = blockIdx.x * 256 + threadIdx.x;
  if (g >= NSEG) return;
  int lo = 0, hi = N;
  while (lo < hi) { int m = (lo + hi) >> 1; if (batch[m] < g) lo = m + 1; else hi = m; }
  int first = lo;
  lo = 0; hi = N;
  while (lo < hi) { int m = (lo + hi) >> 1; if (batch[m] < g + 1) lo = m + 1; else hi = m; }
  counts[g] = (float)(lo - first);
}

// ---------------- fused GEMM (f16 WMMA, f32 accum) + segment-sum -----------
// Block: 256 threads = 8 waves. Tile: 32 rows x 512 cols.
// Wave w: m-half = w&1 (rows m*16..m*16+15), column quarter cq = w>>1
// (8 n-tiles of 16 cols). 8 K-steps of 32 over IN=256.
__global__ __launch_bounds__(256)
void fused_gemm_pool(const float* __restrict__ x,
                     const _Float16* __restrict__ frag,
                     const int* __restrict__ batch,
                     float* __restrict__ sums, int N) {
  __shared__ _Float16 xs[MTILE][40];   // 32 rows x 32 halves, padded (80B rows)
  __shared__ int segs[MTILE];

  const int tid  = threadIdx.x;
  const int lane = tid & 31;
  const int w    = tid >> 5;
  const int m    = w & 1;
  const int cq   = w >> 1;
  const int row0 = blockIdx.x * MTILE;

  if (tid < MTILE) {
    int r = row0 + tid;
    segs[tid] = (r < N) ? batch[r] : -1;
  }

  v8f acc[8] = {};

  const int ldr = tid >> 3;   // staging row 0..31
  const int ldq = tid & 7;    // staging quad 0..7 (float4)

  for (int t = 0; t < 8; ++t) {
    __syncthreads();
    {   // stage 32x32 fp32 tile of x -> f16 in LDS (coalesced b128 loads)
      int gr = row0 + ldr;
      v4f v = {};
      if (gr < N)
        v = *(const v4f*)(x + (size_t)gr * IN_DIM + t * 32 + ldq * 4);
      #pragma unroll
      for (int j = 0; j < 4; ++j) xs[ldr][ldq * 4 + j] = (_Float16)v[j];
    }
    __syncthreads();

    // A fragment (16x32 f16): lane L holds row m*16 + (L&15);
    // halves 0..15 -> K = (L>>4)*16 + h (contiguous, matching packed B).
    const int mrow = m * 16 + (lane & 15);
    const int kb   = (lane >> 4) * 16;
    v8h a0 = *(const v8h*)&xs[mrow][kb];
    v8h a1 = *(const v8h*)&xs[mrow][kb + 8];
    v16h a;
    #pragma unroll
    for (int j = 0; j < 8; ++j) { a[j] = a0[j]; a[8 + j] = a1[j]; }

    #pragma unroll
    for (int nt = 0; nt < 8; ++nt) {
      const int c = cq * 8 + nt;
      const v16h bv = *(const v16h*)(frag + ((size_t)(c * 8 + t) * 32 + lane) * 16);
      acc[nt] = __builtin_amdgcn_wmma_f32_16x16x32_f16(
          false, a, false, bv, (short)0, acc[nt], false, false);
    }
  }
  __syncthreads();

  // C layout: vgpr r, lanes 0-15: M=r, N=lane; lanes 16-31: M=8+r, N=lane-16.
  const bool uniform = (segs[0] == segs[MTILE - 1]) && (segs[0] >= 0);
  const int  colb    = (cq * 8) * 16 + (lane & 15);

  if (uniform) {
    const int seg = segs[0];
    #pragma unroll
    for (int nt = 0; nt < 8; ++nt) {
      float s = 0.f;
      #pragma unroll
      for (int r = 0; r < 8; ++r) s += acc[nt][r];
      s += __shfl_xor(s, 16, 32);              // combine M halves across lanes
      if (lane < 16)
        atomicAdd(&sums[(size_t)seg * NCOL + colb + nt * 16], s);
    }
  } else {
    const int rbase = m * 16 + ((lane >> 4) << 3);
    #pragma unroll
    for (int nt = 0; nt < 8; ++nt) {
      const int col = colb + nt * 16;
      #pragma unroll
      for (int r = 0; r < 8; ++r) {
        int seg = segs[rbase + r];
        if (seg >= 0) atomicAdd(&sums[(size_t)seg * NCOL + col], acc[nt][r]);
      }
    }
  }
}

// ---------------- finalize: mean + bias (empty segments -> 0) --------------
__global__ void finalize(const float* __restrict__ sums,
                         const float* __restrict__ counts,
                         const float* __restrict__ bias,
                         float* __restrict__ out) {
  int i = blockIdx.x * 256 + threadIdx.x;
  if (i >= NSEG * NCOL) return;
  int g = i >> 9;
  int c = i & (NCOL - 1);
  float cnt = counts[g];
  out[i] = (cnt > 0.f) ? sums[i] / cnt + bias[c] : 0.f;
}

extern "C" void kernel_launch(void* const* d_in, const int* in_sizes, int n_in,
                              void* d_out, int out_size, void* d_ws, size_t ws_size,
                              hipStream_t stream) {
  const float* x     = (const float*)d_in[0];
  const float* W     = (const float*)d_in[1];
  const float* b     = (const float*)d_in[2];
  const int*   batch = (const int*)d_in[3];
  const int N = in_sizes[0] / IN_DIM;

  float*     sums   = (float*)d_ws;                       // 4096*512 f32 (8 MB)
  float*     counts = sums + (size_t)NSEG * NCOL;         // 4096 f32
  _Float16*  frag   = (_Float16*)(counts + NSEG);         // 131072 f16 (256 KB)

  const int nz = NSEG * NCOL + NSEG;
  zero_ws<<<(nz + 255) / 256, 256, 0, stream>>>(sums, nz);
  pack_w<<<(32 * 8 * 32 + 255) / 256, 256, 0, stream>>>(W, frag);
  seg_counts<<<(NSEG + 255) / 256, 256, 0, stream>>>(batch, counts, N);

  const int ntiles = (N + MTILE - 1) / MTILE;             // 31250 for N=1e6
  fused_gemm_pool<<<ntiles, 256, 0, stream>>>(x, frag, batch, sums, N);

  finalize<<<(NSEG * NCOL + 255) / 256, 256, 0, stream>>>(
      sums, counts, b, (float*)d_out);
}
